// StyleGAN2Generator_31293131719006
// MI455X (gfx1250) — matrix operand order repaired and checked by reading, not compiled, not run on hardware
//
#include <hip/hip_runtime.h>
#include <hip/hip_bf16.h>

// ---------------------------------------------------------------------------
// StyleGAN2 generator forward for gfx1250 (MI455X).
// Heavy 3x3 modulated convs run as implicit GEMM on v_wmma_f32_16x16x32_f16
// with an LDS-staged, double-buffered 64(o) x 64(pixel) macro-tile per
// 8-wave block, 64-wide K slabs (two WMMA K-substeps per barrier).
// Strategy: conv(x, scale*w*s*d) == d[b,o] * conv3x3(x * s[b,c], scale*w)
//   - modulate kernel: (optional bilinear 2x up) * s[c], f32->f16,
//     channel-last layout with 1-pixel zero halo => taps are pointer shifts.
//   - weight prep: f32 [O][C][3][3] -> f16 [O][tap][C] (conv scale folded in)
//     => A-pointer advances linearly (+64/slab) across all 9 taps.
//   - B-pointer also advances linearly; only a row-wrap fix every 3 taps.
//   - WMMA epilogue: * demod + nw*noise + bias, leaky(0.2)
// Small GEMVs (mapping MLP, style, demod, to_rgb with M=3) stay on VALU.
// ---------------------------------------------------------------------------

typedef __attribute__((ext_vector_type(16))) _Float16 v16h;
typedef __attribute__((ext_vector_type(8)))  _Float16 v8h;
typedef __attribute__((ext_vector_type(8)))  float    v8f;

#define LRELU(v) ((v) > 0.0f ? (v) : 0.2f * (v))

__device__ __forceinline__ int iclamp(int v, int lo, int hi) {
    return v < lo ? lo : (v > hi ? hi : v);
}

// ---------------------------------------------------------------------------
// 1) Mapping network: PixelNorm + 8x (EqualLinear(512->512, lr_mul=0.01) + lrelu 0.2)
// ---------------------------------------------------------------------------
__global__ void mapping_kernel(const float* __restrict__ z,
                               const float* __restrict__ map_w,
                               const float* __restrict__ map_b,
                               float* __restrict__ wl) {
    __shared__ float xs[2][512];
    __shared__ float ssum[2];
    int j = threadIdx.x;  // 0..511
    if (j < 2) ssum[j] = 0.0f;
    __syncthreads();
    float z0 = z[j], z1 = z[512 + j];
    atomicAdd(&ssum[0], z0 * z0);
    atomicAdd(&ssum[1], z1 * z1);
    __syncthreads();
    float inv0 = rsqrtf(ssum[0] * (1.0f / 512.0f) + 1e-8f);
    float inv1 = rsqrtf(ssum[1] * (1.0f / 512.0f) + 1e-8f);
    xs[0][j] = z0 * inv0;
    xs[1][j] = z1 * inv1;
    __syncthreads();
    const float scale = 0.01f * 0.044194173824159216f;  // 0.01 / sqrt(512)
    for (int l = 0; l < 8; ++l) {
        const float* w = map_w + (l * 512 + j) * 512;  // row j of layer l
        float a0 = 0.0f, a1 = 0.0f;
        for (int k = 0; k < 512; ++k) {
            float wv = w[k];
            a0 += xs[0][k] * wv;
            a1 += xs[1][k] * wv;
        }
        float bb = map_b[l * 512 + j] * 0.01f;
        a0 = LRELU(a0 * scale + bb);
        a1 = LRELU(a1 * scale + bb);
        __syncthreads();
        xs[0][j] = a0;
        xs[1][j] = a1;
        __syncthreads();
    }
    wl[j]       = xs[0][j];
    wl[512 + j] = xs[1][j];
}

// ---------------------------------------------------------------------------
// 2) Style: s[b,i] = (wl[b] . mw[i]) / sqrt(512) + mb[i],  i in [0,C)
// ---------------------------------------------------------------------------
__global__ void style_kernel(const float* __restrict__ wl,
                             const float* __restrict__ mw,
                             const float* __restrict__ mb,
                             float* __restrict__ s, int C) {
    int i = blockIdx.x * blockDim.x + threadIdx.x;
    if (i >= C) return;
    const float scale = 0.044194173824159216f;  // 1/sqrt(512)
    const float* row = mw + (long)i * 512;
    float a0 = 0.0f, a1 = 0.0f;
    for (int k = 0; k < 512; ++k) {
        float w = row[k];
        a0 += wl[k] * w;
        a1 += wl[512 + k] * w;
    }
    float b = mb[i];
    s[i]     = a0 * scale + b;
    s[C + i] = a1 * scale + b;
}

// ---------------------------------------------------------------------------
// 3) Demod: d[b,o] = rsqrt( sum_{c,k} (scale*w[o,c,k]*s[b,c])^2 + 1e-8 )
// ---------------------------------------------------------------------------
__global__ void demod_kernel(const float* __restrict__ w,
                             const float* __restrict__ s,
                             float* __restrict__ dcoef,
                             int C, int O, float scale) {
    int idx = blockIdx.x * blockDim.x + threadIdx.x;  // b*O + o
    if (idx >= 2 * O) return;
    int b = idx / O, o = idx % O;
    const float* wo = w + (long)o * C * 9;
    const float* sb = s + b * C;
    float sum = 0.0f;
    for (int c = 0; c < C; ++c) {
        float sv = sb[c] * scale;
        const float* wc = wo + c * 9;
        float t = 0.0f;
#pragma unroll
        for (int k = 0; k < 9; ++k) { float v = wc[k]; t += v * v; }
        sum += t * sv * sv;
    }
    dcoef[idx] = rsqrtf(sum + 1e-8f);
}

// ---------------------------------------------------------------------------
// 4) Weight prep: f32 [O][C][3][3] -> f16 [O][tap][C], conv scale folded in.
// ---------------------------------------------------------------------------
__global__ void wprep_kernel(const float* __restrict__ w,
                             _Float16* __restrict__ wA,
                             int C, int O, float scale) {
    long idx = (long)blockIdx.x * blockDim.x + threadIdx.x;
    long total = (long)O * C * 9;
    if (idx >= total) return;
    int c = (int)(idx % C);
    long rest = idx / C;
    int t = (int)(rest % 9);
    int o = (int)(rest / 9);
    wA[idx] = (_Float16)(w[((long)o * C + c) * 9 + t] * scale);
}

// ---------------------------------------------------------------------------
// 5) Modulate (+optional bilinear 2x upsample, half-pixel centers, clamped):
//    xm[b][y+1][x+1][c] = f16( up(x)[b,c,y,x] * s[b,c] ), halo = 0.
// ---------------------------------------------------------------------------
__global__ void modulate_kernel(const float* __restrict__ xin,
                                const float* __restrict__ s,
                                _Float16* __restrict__ xm,
                                int C, int R, int up, long in_bstride) {
    long idx = (long)blockIdx.x * blockDim.x + threadIdx.x;
    int P = R + 2;
    long total = 2L * P * P * C;
    if (idx >= total) return;
    int c = (int)(idx % C);
    long rest = idx / C;
    int xq = (int)(rest % P); rest /= P;
    int yq = (int)(rest % P);
    int b  = (int)(rest / P);
    int y = yq - 1, x = xq - 1;
    float v = 0.0f;
    if (y >= 0 && y < R && x >= 0 && x < R) {
        int Rin = up ? (R >> 1) : R;
        const float* xb = xin + (long)b * in_bstride + (long)c * Rin * Rin;
        if (!up) {
            v = xb[y * R + x];
        } else {
            float sy = (y + 0.5f) * 0.5f - 0.5f;
            float sx = (x + 0.5f) * 0.5f - 0.5f;
            int y0 = (int)floorf(sy); float fy = sy - (float)y0;
            int x0 = (int)floorf(sx); float fx = sx - (float)x0;
            int y0c = iclamp(y0, 0, Rin - 1), y1c = iclamp(y0 + 1, 0, Rin - 1);
            int x0c = iclamp(x0, 0, Rin - 1), x1c = iclamp(x0 + 1, 0, Rin - 1);
            float v00 = xb[y0c * Rin + x0c], v01 = xb[y0c * Rin + x1c];
            float v10 = xb[y1c * Rin + x0c], v11 = xb[y1c * Rin + x1c];
            v = v00 * (1.0f - fy) * (1.0f - fx) + v01 * (1.0f - fy) * fx +
                v10 * fy * (1.0f - fx) + v11 * fy * fx;
        }
        v *= s[b * C + c];
    }
    xm[idx] = (_Float16)v;
}

// ---------------------------------------------------------------------------
// 6) Implicit-GEMM 3x3 conv on WMMA f16 (f32 accum), LDS macro-tile version.
//    Block = 256 threads (8 waves): 64(o) x 64(pixel) tile, K slab = 64.
//    Waves: wo = wave>>1 selects o-subtile, wp = wave&1 selects pixel half;
//    each wave owns two 16x16 accumulators (2 pixel tiles) and issues
//    4 WMMAs per barrier (2 K-substeps x 2 tiles).
//    Staging pointers advance linearly: A is contiguous across taps by
//    construction; B needs only a row-wrap fix every 3 taps (no division).
//    Fragment placement: contiguous K per lane half (K 16h..16h+15), same order for A and B.
// ---------------------------------------------------------------------------
__global__ __launch_bounds__(256) void conv_wmma_kernel(
    const _Float16* __restrict__ wA,
    const _Float16* __restrict__ xm,
    const float* __restrict__ dcoef,
    const float* __restrict__ noise,
    const float* __restrict__ nw,
    const float* __restrict__ bias,
    float* __restrict__ out,
    int C, int O, int R) {
    // Row stride 72 halves (144B = 9*16B): 16B alignment, distinct bank
    // groups for 16 consecutive rows (36*n mod 64 has period 16).
    __shared__ _Float16 As[2][64][72];  // [buf][o_local][k0..63]
    __shared__ _Float16 Bs[2][64][72];  // [buf][pix_local][k0..63]

    const int lane = threadIdx.x;                 // 0..31
    const int wv   = threadIdx.y;                 // 0..7
    const int tid  = wv * 32 + lane;              // 0..255
    const int wo   = wv >> 1;                     // 0..3 : o subtile
    const int wp   = wv & 1;                      // 0..1 : pixel half
    const int b    = blockIdx.z;
    const int P    = R + 2;
    const int N    = R * R;
    const bool hi  = lane >= 16;
    const int l15  = lane & 15;

    // ---- staging-thread mapping: 256 threads x 32B cover an 8KB slab ----
    const int srow   = tid >> 2;        // 0..63
    const int schunk = (tid & 3) * 16;  // halves: 0,16,32,48

    const int aorow = blockIdx.y * 64 + srow;     // O is a multiple of 64
    const _Float16* a_next = wA + (long)aorow * 9 * C + schunk;

    int sp = blockIdx.x * 64 + srow;
    if (sp > N - 1) sp = N - 1;                   // clamp for tiny layers
    const int spy = sp / R, spx = sp % R;
    const _Float16* b_next = xm + (((long)b * P + spy) * P + spx) * C + schunk;

    // ---- compute-side fragment indices ----
    const int arow  = wo * 16 + l15;    // A local row
    const int akoff = hi ? 16 : 0;      // halves: A K base (K 0..15 / 16..31)
    const int bkoff = hi ? 16 : 0;      // halves: B K base (K 0..15 / 16..31)
    const int bn0   = wp * 32 + l15;    // local pixel rows of the 2 tiles
    const int bn1   = bn0 + 16;
    const int p0    = blockIdx.x * 64 + bn0;
    const int p1    = p0 + 16;

    v8f acc0 = {}, acc1 = {};

    const int  slabsPerTap = C >> 6;              // 64-K slabs per tap
    const int  stepsPerRow = 3 * slabsPerTap;     // slabs per kernel row
    const int  nsteps      = 9 * slabsPerTap;
    const long rowfix      = (long)(P - 3) * C;   // B jump at kernel-row wrap
    int cleft = stepsPerRow;

    // stage slab 0 (tap 0, c0 0)
    *(v8h*)&As[0][srow][schunk]     = *(const v8h*)(a_next);
    *(v8h*)&As[0][srow][schunk + 8] = *(const v8h*)(a_next + 8);
    *(v8h*)&Bs[0][srow][schunk]     = *(const v8h*)(b_next);
    *(v8h*)&Bs[0][srow][schunk + 8] = *(const v8h*)(b_next + 8);
    a_next += 64;
    b_next += 64;
    if (--cleft == 0) { b_next += rowfix; cleft = stepsPerRow; }
    __syncthreads();

    for (int step = 0; step < nsteps; ++step) {
        const int db = step & 1;
        if (step + 1 < nsteps) {        // prefetch next slab into other buffer
            *(v8h*)&As[db ^ 1][srow][schunk]     = *(const v8h*)(a_next);
            *(v8h*)&As[db ^ 1][srow][schunk + 8] = *(const v8h*)(a_next + 8);
            *(v8h*)&Bs[db ^ 1][srow][schunk]     = *(const v8h*)(b_next);
            *(v8h*)&Bs[db ^ 1][srow][schunk + 8] = *(const v8h*)(b_next + 8);
            a_next += 64;
            b_next += 64;
            if (--cleft == 0) { b_next += rowfix; cleft = stepsPerRow; }
        }
        // two 32-K substeps from LDS buffer db
#pragma unroll
        for (int sub = 0; sub < 2; ++sub) {
            const int ka = sub * 32 + akoff;
            const int kb = sub * 32 + bkoff;
            v8h alo = *(const v8h*)&As[db][arow][ka];       // K 0..7  (+16 hi)
            v8h ahi = *(const v8h*)&As[db][arow][ka + 8];   // K 8..15 (+16 hi)
            v8h x0  = *(const v8h*)&Bs[db][bn0][kb];        // K 0..7  (+16 hi)
            v8h x1  = *(const v8h*)&Bs[db][bn0][kb + 8];    // K 8..15 (+16 hi)
            v8h y0  = *(const v8h*)&Bs[db][bn1][kb];
            v8h y1  = *(const v8h*)&Bs[db][bn1][kb + 8];
            v16h a  = __builtin_shufflevector(alo, ahi, 0, 1, 2, 3, 4, 5, 6, 7,
                                              8, 9, 10, 11, 12, 13, 14, 15);
            v16h b0 = __builtin_shufflevector(x0, x1, 0, 1, 2, 3, 4, 5, 6, 7,
                                              8, 9, 10, 11, 12, 13, 14, 15);
            v16h b1 = __builtin_shufflevector(y0, y1, 0, 1, 2, 3, 4, 5, 6, 7,
                                              8, 9, 10, 11, 12, 13, 14, 15);
            acc0 = __builtin_amdgcn_wmma_f32_16x16x32_f16(
                false, a, false, b0, (short)0, acc0, false, false);
            acc1 = __builtin_amdgcn_wmma_f32_16x16x32_f16(
                false, a, false, b1, (short)0, acc1, false, false);
        }
        __syncthreads();  // buffer db consumed; db^1 staged for next step
    }

    // ---- epilogue: demod, noise, bias, leaky 0.2 ----
    const float nwv = nw[0];
    if (p0 < N) {  // N is a multiple of 16 => uniform across the tile
        const int py = p0 / R, px = p0 % R;
        const float noi = noise[((long)b * R + py) * R + px] * nwv;
#pragma unroll
        for (int r = 0; r < 8; ++r) {
            int oo = blockIdx.y * 64 + wo * 16 + (hi ? 8 + r : r);
            float v = acc0[r] * dcoef[b * O + oo] + noi + bias[oo];
            v = LRELU(v);
            out[(((long)b * O + oo) * R + py) * R + px] = v;
        }
    }
    if (p1 < N) {
        const int py = p1 / R, px = p1 % R;
        const float noi = noise[((long)b * R + py) * R + px] * nwv;
#pragma unroll
        for (int r = 0; r < 8; ++r) {
            int oo = blockIdx.y * 64 + wo * 16 + (hi ? 8 + r : r);
            float v = acc1[r] * dcoef[b * O + oo] + noi + bias[oo];
            v = LRELU(v);
            out[(((long)b * O + oo) * R + py) * R + px] = v;
        }
    }
}

// ---------------------------------------------------------------------------
// 7) to_rgb: 1x1 modulated conv (no demod, M=3) + bias + upsampled skip add.
// ---------------------------------------------------------------------------
__global__ void rgb_kernel(const float* __restrict__ x,
                           const float* __restrict__ s,
                           const float* __restrict__ w,
                           const float* __restrict__ bias,
                           const float* __restrict__ skip_in,
                           float* __restrict__ skip_out,
                           int C, int R, int has_skip) {
    int idx = blockIdx.x * blockDim.x + threadIdx.x;  // over 2*R*R
    if (idx >= 2 * R * R) return;
    int b = idx / (R * R);
    int p = idx % (R * R);
    int y = p / R, xq = p % R;
    float scale = rsqrtf((float)C);
    const float* xb = x + (long)b * C * R * R + p;
    const float* sb = s + b * C;
    float a0 = 0.0f, a1 = 0.0f, a2 = 0.0f;
    for (int c = 0; c < C; ++c) {
        float v = xb[(long)c * R * R] * sb[c];
        a0 += v * w[c];
        a1 += v * w[C + c];
        a2 += v * w[2 * C + c];
    }
    a0 = a0 * scale + bias[0];
    a1 = a1 * scale + bias[1];
    a2 = a2 * scale + bias[2];
    if (has_skip) {
        int Rin = R >> 1;
        float sy = (y + 0.5f) * 0.5f - 0.5f;
        float sx = (xq + 0.5f) * 0.5f - 0.5f;
        int y0 = (int)floorf(sy); float fy = sy - (float)y0;
        int x0 = (int)floorf(sx); float fx = sx - (float)x0;
        int y0c = iclamp(y0, 0, Rin - 1), y1c = iclamp(y0 + 1, 0, Rin - 1);
        int x0c = iclamp(x0, 0, Rin - 1), x1c = iclamp(x0 + 1, 0, Rin - 1);
        float w00 = (1.0f - fy) * (1.0f - fx), w01 = (1.0f - fy) * fx;
        float w10 = fy * (1.0f - fx),          w11 = fy * fx;
#pragma unroll
        for (int ch = 0; ch < 3; ++ch) {
            const float* sk = skip_in + ((long)b * 3 + ch) * Rin * Rin;
            float u = sk[y0c * Rin + x0c] * w00 + sk[y0c * Rin + x1c] * w01 +
                      sk[y1c * Rin + x0c] * w10 + sk[y1c * Rin + x1c] * w11;
            if (ch == 0) a0 += u; else if (ch == 1) a1 += u; else a2 += u;
        }
    }
    long ob = (long)b * 3 * R * R + p;
    skip_out[ob]             = a0;
    skip_out[ob + R * R]     = a1;
    skip_out[ob + 2 * R * R] = a2;
}

// ---------------------------------------------------------------------------
// Host orchestration
// ---------------------------------------------------------------------------
namespace {

struct ConvP { const float *w, *mw, *mb, *nw, *b; };
struct RgbP  { const float *w, *mw, *mb, *b; };
struct Scratch {
    float *wl, *sbuf, *dbuf, *act, *skipA, *skipB;
    _Float16 *wA, *xm;
};

inline size_t alup(size_t x) { return (x + 255) & ~(size_t)255; }

void run_styled(const ConvP& p, const float* noise, const float* xin, long in_bstride,
                float* xout, int C, int O, int R, int up, const Scratch& ws,
                hipStream_t stream) {
    const float conv_scale = rsqrtf((float)(C * 9));
    // style s[b, C]
    style_kernel<<<(C + 255) / 256, 256, 0, stream>>>(ws.wl, p.mw, p.mb, ws.sbuf, C);
    // demod d[b, O]
    demod_kernel<<<(2 * O + 255) / 256, 256, 0, stream>>>(p.w, ws.sbuf, ws.dbuf, C, O, conv_scale);
    // weight prep f16 [O][9][C]
    long wtot = (long)O * C * 9;
    wprep_kernel<<<(int)((wtot + 255) / 256), 256, 0, stream>>>(p.w, ws.wA, C, O, conv_scale);
    // modulated padded f16 input [2][R+2][R+2][C]
    long mtot = 2L * (R + 2) * (R + 2) * C;
    modulate_kernel<<<(int)((mtot + 255) / 256), 256, 0, stream>>>(
        xin, ws.sbuf, ws.xm, C, R, up, in_bstride);
    // WMMA conv + epilogue: 64x64 macro-tiles, 8 waves/block
    dim3 grid((R * R + 63) / 64, O / 64, 2), blk(32, 8, 1);
    conv_wmma_kernel<<<grid, blk, 0, stream>>>(ws.wA, ws.xm, ws.dbuf, noise, p.nw, p.b,
                                               xout, C, O, R);
}

void run_rgb(const RgbP& p, const float* x, const float* skip_in, float* skip_out,
             int C, int R, int has_skip, const Scratch& ws, hipStream_t stream) {
    style_kernel<<<(C + 255) / 256, 256, 0, stream>>>(ws.wl, p.mw, p.mb, ws.sbuf, C);
    rgb_kernel<<<(2 * R * R + 255) / 256, 256, 0, stream>>>(
        x, ws.sbuf, p.w, p.b, skip_in, skip_out, C, R, has_skip);
}

}  // namespace

extern "C" void kernel_launch(void* const* d_in, const int* in_sizes, int n_in,
                              void* d_out, int out_size, void* d_ws, size_t ws_size,
                              hipStream_t stream) {
    (void)in_sizes; (void)n_in; (void)out_size; (void)ws_size;

    const float* z      = (const float*)d_in[0];
    const float* map_w  = (const float*)d_in[1];
    const float* map_b  = (const float*)d_in[2];
    const float* constx = (const float*)d_in[3];

    ConvP cp[11];
    for (int L = 0; L < 11; ++L) {
        cp[L].w  = (const float*)d_in[4 + 5 * L + 0];
        cp[L].mw = (const float*)d_in[4 + 5 * L + 1];
        cp[L].mb = (const float*)d_in[4 + 5 * L + 2];
        cp[L].nw = (const float*)d_in[4 + 5 * L + 3];
        cp[L].b  = (const float*)d_in[4 + 5 * L + 4];
    }
    const float* noises[11];
    for (int L = 0; L < 11; ++L) noises[L] = (const float*)d_in[59 + L];
    RgbP rp[6];
    for (int i = 0; i < 6; ++i) {
        rp[i].w  = (const float*)d_in[70 + 4 * i + 0];
        rp[i].mw = (const float*)d_in[70 + 4 * i + 1];
        rp[i].mb = (const float*)d_in[70 + 4 * i + 2];
        rp[i].b  = (const float*)d_in[70 + 4 * i + 3];
    }

    // Workspace bump allocation
    char* base = (char*)d_ws;
    size_t off = 0;
    Scratch ws;
    ws.wl    = (float*)(base + off);    off = alup(off + 1024 * sizeof(float));
    ws.sbuf  = (float*)(base + off);    off = alup(off + 1024 * sizeof(float));
    ws.dbuf  = (float*)(base + off);    off = alup(off + 1024 * sizeof(float));
    ws.wA    = (_Float16*)(base + off); off = alup(off + (size_t)512 * 9 * 512 * sizeof(_Float16));
    ws.xm    = (_Float16*)(base + off); off = alup(off + (size_t)2 * 130 * 130 * 512 * sizeof(_Float16));
    ws.act   = (float*)(base + off);    off = alup(off + (size_t)2 * 256 * 128 * 128 * sizeof(float));
    ws.skipA = (float*)(base + off);    off = alup(off + (size_t)2 * 3 * 128 * 128 * sizeof(float));
    ws.skipB = (float*)(base + off);    off = alup(off + (size_t)2 * 3 * 128 * 128 * sizeof(float));

    // Layer specs: {C_in, C_out, out_res, upsample}
    struct LS { int C, O, R, up; };
    const LS ls[11] = {{512,512,4,0},
                       {512,512,8,1},{512,512,8,0},
                       {512,512,16,1},{512,512,16,0},
                       {512,512,32,1},{512,512,32,0},
                       {512,512,64,1},{512,512,64,0},
                       {512,256,128,1},{256,256,128,0}};

    // 1) mapping -> wl [2,512]
    mapping_kernel<<<1, 512, 0, stream>>>(z, map_w, map_b, ws.wl);

    // 2) conv0 from tiled const (batch stride 0) -> act
    run_styled(cp[0], noises[0], constx, /*in_bstride=*/0, ws.act,
               ls[0].C, ls[0].O, ls[0].R, ls[0].up, ws, stream);

    // 3) rgb0 -> skipA
    run_rgb(rp[0], ws.act, nullptr, ws.skipA, 512, 4, /*has_skip=*/0, ws, stream);

    float* skip_cur = ws.skipA;
    float* skip_nxt = ws.skipB;

    // 4) 5 blocks of (up-conv, conv, to_rgb)
    for (int i = 0; i < 5; ++i) {
        const LS& a = ls[1 + 2 * i];  // upsample conv
        const LS& c = ls[2 + 2 * i];  // plain conv
        long bs_a = (long)a.C * (a.R / 2) * (a.R / 2);
        run_styled(cp[1 + 2 * i], noises[1 + 2 * i], ws.act, bs_a, ws.act,
                   a.C, a.O, a.R, 1, ws, stream);
        long bs_c = (long)c.C * c.R * c.R;
        run_styled(cp[2 + 2 * i], noises[2 + 2 * i], ws.act, bs_c, ws.act,
                   c.C, c.O, c.R, 0, ws, stream);
        float* dst = (i == 4) ? (float*)d_out : skip_nxt;
        run_rgb(rp[1 + i], ws.act, skip_cur, dst, c.O, c.R, /*has_skip=*/1, ws, stream);
        float* t = skip_cur; skip_cur = skip_nxt; skip_nxt = t;
    }
}
